// Net_72825465471121
// MI455X (gfx1250) — compile-verified, matrix operand order checked by reading, not run on hardware
//
#include <hip/hip_runtime.h>
#include <hip/hip_bf16.h>

// ---------------------------------------------------------------------------
// CDNA5 (gfx1250) 3-level deep-homography network.
// Conv/FC run through v_wmma_f32_16x16x32_f16 with branch-free inner loops:
//  - convs: weights + im2col pre-packed to f16 in exact WMMA fragment order,
//           2x2 register-tiled GEMM (4 WMMAs per 4 fragment loads)
//  - FCs:   direct float4 loads (K always a multiple of 32), row-clamped
// ---------------------------------------------------------------------------

typedef __attribute__((ext_vector_type(16))) _Float16 v16h;
typedef __attribute__((ext_vector_type(8)))  float    v8f;

#define SLOPE 0.01f
#define IMGF  128.0f

// Fragment element -> K offset (16-bit A 16x32 layout, ISA 7.12.2):
//   lane L: row = L&15, kb = (L<16)?0:8 ; elem j<8 -> K=kb+j ; j>=8 -> K=kb+j+8
__device__ __forceinline__ int frag_k(int kb, int j) {
    return kb + ((j < 8) ? j : (j + 8));
}

// ---------------------------------------------------------------------------
// Pack conv weights [Cout][Cin*9] -> Apack f16 [mt][kt][lane][16] (zero pad)
// ---------------------------------------------------------------------------
__global__ void pack_w_k(const float* __restrict__ w, _Float16* __restrict__ Ap,
                         int Mtiles, int KT, int Cout, int K)
{
    const long total = (long)Mtiles * KT * 512;
    long t = (long)blockIdx.x * blockDim.x + threadIdx.x;
    if (t >= total) return;
    const int j    = t & 15;
    const int lane = (t >> 4) & 31;
    const long rest = t >> 9;
    const int kt = (int)(rest % KT);
    const int mt = (int)(rest / KT);
    const int m  = mt * 16 + (lane & 15);
    const int kb = (lane < 16) ? 0 : 8;
    const int k  = kt * 32 + frag_k(kb, j);
    float v = 0.0f;
    if (m < Cout && k < K) v = w[(long)m * K + k];
    Ap[t] = (_Float16)v;
}

// ---------------------------------------------------------------------------
// Pack im2col patches -> Bpack f16 [nt][kt][lane][16] for a chunk of N tiles.
//   n = nBase + nt*16 + (lane&15) indexes flattened (b,y,x); k -> (ci,ky,kx).
// ---------------------------------------------------------------------------
__global__ void pack_b_k(const float* __restrict__ in, _Float16* __restrict__ Bp,
                         int ctiles, int KT, int nBase,
                         int Cin, int H, int W, int K, int Npix)
{
    const long total = (long)ctiles * KT * 512;
    long t = (long)blockIdx.x * blockDim.x + threadIdx.x;
    if (t >= total) return;
    const int j    = t & 15;
    const int lane = (t >> 4) & 31;
    const long rest = t >> 9;
    const int kt = (int)(rest % KT);
    const int nt = (int)(rest / KT);
    const int kb = (lane < 16) ? 0 : 8;
    const int k  = kt * 32 + frag_k(kb, j);
    const int n  = nBase + nt * 16 + (lane & 15);
    float v = 0.0f;
    if (k < K && n < Npix) {
        const int HW = H * W;
        const int ci = k / 9;
        const int r9 = k - ci * 9;
        const int ky = r9 / 3;
        const int kx = r9 - ky * 3;
        const int b  = n / HW;
        const int rem = n - b * HW;
        const int oy = rem / W;
        const int ox = rem - oy * W;
        const int iy = oy + ky - 1;
        const int ix = ox + kx - 1;
        if (iy >= 0 && iy < H && ix >= 0 && ix < W)
            v = in[((b * Cin + ci) * H + iy) * W + ix];
    }
    Bp[t] = (_Float16)v;
}

// ---------------------------------------------------------------------------
// 2x2 register-tiled packed GEMM.
// Wave owns a 32(M) x 32(N) macro-tile = 4 accumulators; per K-step it loads
// 4 fragments (2 A + 2 B, 32B/lane each, wave-contiguous 1KB) and issues
// 4 WMMAs -> 16 FLOP/byte from L2 (2x the untiled version).
// Block = 128 threads = 4 waves; grid = (Mtiles/2, ctiles/8).
// ---------------------------------------------------------------------------
__device__ __forceinline__ void store_tile(const v8f& acc, const float* bias,
                                           float* out, int mt, int ntile,
                                           int lane, int Cout, int HW, int Npix,
                                           int nBase, int relu)
{
    const int coBase = mt * 16 + ((lane < 16) ? 0 : 8);
    const int n = nBase + ntile * 16 + (lane & 15);
    if (n < Npix) {
        const int b2   = n / HW;
        const int rem2 = n - b2 * HW;
#pragma unroll
        for (int r = 0; r < 8; ++r) {
            const int co = coBase + r;
            if (co < Cout) {
                float v = acc[r] + bias[co];
                if (relu) v = v > 0.0f ? v : 0.0f;
                out[(b2 * Cout + co) * HW + rem2] = v;
            }
        }
    }
}

__global__ void conv_gemm_k(const _Float16* __restrict__ Ap,
                            const _Float16* __restrict__ Bp,
                            const float* __restrict__ bias,
                            float* __restrict__ out,
                            int KT, int Cout, int HW, int Npix,
                            int nBase, int relu)
{
    const int lane = threadIdx.x & 31;
    const int wav  = threadIdx.x >> 5;
    const int mt0  = blockIdx.x * 2;
    const int nt0  = (blockIdx.y * 4 + wav) * 2;

    const _Float16* ap0 = Ap + (long)(mt0 + 0) * KT * 512 + lane * 16;
    const _Float16* ap1 = Ap + (long)(mt0 + 1) * KT * 512 + lane * 16;
    const _Float16* bp0 = Bp + (long)(nt0 + 0) * KT * 512 + lane * 16;
    const _Float16* bp1 = Bp + (long)(nt0 + 1) * KT * 512 + lane * 16;

    v8f acc00 = {}, acc01 = {}, acc10 = {}, acc11 = {};
    for (int kt = 0; kt < KT; ++kt) {
        v16h a0 = *(const v16h*)ap0;
        v16h a1 = *(const v16h*)ap1;
        v16h b0 = *(const v16h*)bp0;
        v16h b1 = *(const v16h*)bp1;
        ap0 += 512; ap1 += 512; bp0 += 512; bp1 += 512;
        acc00 = __builtin_amdgcn_wmma_f32_16x16x32_f16(false, a0, false, b0, (short)0, acc00, false, false);
        acc01 = __builtin_amdgcn_wmma_f32_16x16x32_f16(false, a0, false, b1, (short)0, acc01, false, false);
        acc10 = __builtin_amdgcn_wmma_f32_16x16x32_f16(false, a1, false, b0, (short)0, acc10, false, false);
        acc11 = __builtin_amdgcn_wmma_f32_16x16x32_f16(false, a1, false, b1, (short)0, acc11, false, false);
    }

    store_tile(acc00, bias, out, mt0 + 0, nt0 + 0, lane, Cout, HW, Npix, nBase, relu);
    store_tile(acc01, bias, out, mt0 + 0, nt0 + 1, lane, Cout, HW, Npix, nBase, relu);
    store_tile(acc10, bias, out, mt0 + 1, nt0 + 0, lane, Cout, HW, Npix, nBase, relu);
    store_tile(acc11, bias, out, mt0 + 1, nt0 + 1, lane, Cout, HW, Npix, nBase, relu);
}

// ---------------------------------------------------------------------------
// WMMA fully-connected: out[n][m] = relu(sum_k X[n][k]*Wt[m][k] + b[m]).
// K is always a multiple of 32 here; rows are index-clamped (garbage rows
// only feed D rows/cols that are never stored), so the loop is branch-free:
// 8 float4 loads + cvt + 1 wmma per K-step. This phase streams the 3.8 GB
// of f32 FC weights and is HBM-bandwidth-bound; cvts hide under the loads.
// ---------------------------------------------------------------------------
__global__ void fc_wmma(const float* __restrict__ X,
                        const float* __restrict__ Wt,
                        const float* __restrict__ bias,
                        float* __restrict__ out,
                        int N, long K, int M, int relu)
{
    const int lane  = threadIdx.x & 31;
    const int wav   = threadIdx.x >> 5;
    const int mBase = blockIdx.x * 64 + wav * 16;
    if (mBase >= M) return;                         // wave-uniform

    const int row = lane & 15;
    const int kb  = (lane < 16) ? 0 : 8;
    const int mc  = min(mBase + row, M - 1);
    const int nc  = min(row, N - 1);
    const float* wp = Wt + (long)mc * K + kb;
    const float* xp = X  + (long)nc * K + kb;

    v8f acc = {};
    for (long k0 = 0; k0 < K; k0 += 32) {
        const float4 a0 = *(const float4*)(wp + k0);
        const float4 a1 = *(const float4*)(wp + k0 + 4);
        const float4 a2 = *(const float4*)(wp + k0 + 16);
        const float4 a3 = *(const float4*)(wp + k0 + 20);
        const float4 b0 = *(const float4*)(xp + k0);
        const float4 b1 = *(const float4*)(xp + k0 + 4);
        const float4 b2 = *(const float4*)(xp + k0 + 16);
        const float4 b3 = *(const float4*)(xp + k0 + 20);
        v16h af, bf;
        af[0]  = (_Float16)a0.x; af[1]  = (_Float16)a0.y; af[2]  = (_Float16)a0.z; af[3]  = (_Float16)a0.w;
        af[4]  = (_Float16)a1.x; af[5]  = (_Float16)a1.y; af[6]  = (_Float16)a1.z; af[7]  = (_Float16)a1.w;
        af[8]  = (_Float16)a2.x; af[9]  = (_Float16)a2.y; af[10] = (_Float16)a2.z; af[11] = (_Float16)a2.w;
        af[12] = (_Float16)a3.x; af[13] = (_Float16)a3.y; af[14] = (_Float16)a3.z; af[15] = (_Float16)a3.w;
        bf[0]  = (_Float16)b0.x; bf[1]  = (_Float16)b0.y; bf[2]  = (_Float16)b0.z; bf[3]  = (_Float16)b0.w;
        bf[4]  = (_Float16)b1.x; bf[5]  = (_Float16)b1.y; bf[6]  = (_Float16)b1.z; bf[7]  = (_Float16)b1.w;
        bf[8]  = (_Float16)b2.x; bf[9]  = (_Float16)b2.y; bf[10] = (_Float16)b2.z; bf[11] = (_Float16)b2.w;
        bf[12] = (_Float16)b3.x; bf[13] = (_Float16)b3.y; bf[14] = (_Float16)b3.z; bf[15] = (_Float16)b3.w;
        acc = __builtin_amdgcn_wmma_f32_16x16x32_f16(
            false, af, false, bf, (short)0, acc, false, false);
    }

    const int coBase = mBase + ((lane < 16) ? 0 : 8);
    const int n = lane & 15;
    if (n < N) {
#pragma unroll
        for (int r = 0; r < 8; ++r) {
            const int mm = coBase + r;
            if (mm < M) {
                float v = acc[r] + bias[mm];
                if (relu) v = v > 0.0f ? v : 0.0f;
                out[(long)n * M + mm] = v;
            }
        }
    }
}

// ---------------------------------------------------------------------------
// 2x2 max-pool (stride 2, VALID)
// ---------------------------------------------------------------------------
__global__ void maxpool2_k(const float* __restrict__ in, float* __restrict__ out,
                           int Bn, int C, int H, int W)
{
    const int oh = H >> 1, ow = W >> 1;
    const int total = Bn * C * oh * ow;
    int t = blockIdx.x * blockDim.x + threadIdx.x;
    if (t >= total) return;
    const int x = t % ow;
    const int y = (t / ow) % oh;
    const int c = (t / (ow * oh)) % C;
    const int b = t / (ow * oh * C);
    const float* p = in + ((b * C + c) * H + 2 * y) * W + 2 * x;
    float m0 = fmaxf(p[0], p[1]);
    float m1 = fmaxf(p[W], p[W + 1]);
    out[t] = fmaxf(m0, m1);
}

// ---------------------------------------------------------------------------
// L2 normalize across channels at each (b, pixel)
// ---------------------------------------------------------------------------
__global__ void l2norm_k(const float* __restrict__ in, float* __restrict__ out,
                         int Bn, int C, int HW)
{
    int t = blockIdx.x * blockDim.x + threadIdx.x;
    if (t >= Bn * HW) return;
    const int b = t / HW;
    const int p = t - b * HW;
    float s = 0.0f;
    for (int c = 0; c < C; ++c) {
        float v = in[(b * C + c) * HW + p];
        s += v * v;
    }
    float n = sqrtf(s);
    if (n == 0.0f) n = 1.0f;
    const float inv = 1.0f / n;
    for (int c = 0; c < C; ++c)
        out[(b * C + c) * HW + p] = in[(b * C + c) * HW + p] * inv;
}

// ---------------------------------------------------------------------------
// Correlation cost volume + leaky-relu (zero-padded window)
// ---------------------------------------------------------------------------
__global__ void corr_k(const float* __restrict__ f1, const float* __restrict__ f2,
                       float* __restrict__ out, int Bn, int C, int H, int W, int sr)
{
    const int D = 2 * sr + 1;
    const int total = Bn * D * D * H * W;
    int t = blockIdx.x * blockDim.x + threadIdx.x;
    if (t >= total) return;
    const int x = t % W;
    const int y = (t / W) % H;
    const int d = (t / (W * H)) % (D * D);
    const int b = t / (W * H * D * D);
    const int dy = d / D - sr;
    const int dx = d % D - sr;
    const int y2 = y + dy, x2 = x + dx;
    float s = 0.0f;
    if (y2 >= 0 && y2 < H && x2 >= 0 && x2 < W) {
        const float* a  = f1 + ((long)b * C) * H * W + y * W + x;
        const float* c2 = f2 + ((long)b * C) * H * W + y2 * W + x2;
        for (int c = 0; c < C; ++c)
            s += a[c * H * W] * c2[c * H * W];
        s /= (float)C;
    }
    out[t] = s >= 0.0f ? s : SLOPE * s;
}

// ---------------------------------------------------------------------------
// DLT solve + 3x3 inverse + homography normalization. One thread per batch.
// ---------------------------------------------------------------------------
__global__ void dlt_norm_k(const float* __restrict__ o1, const float* __restrict__ o2,
                           float invScale, float size, float* __restrict__ theta, int Bn)
{
    int b = blockIdx.x * blockDim.x + threadIdx.x;
    if (b >= Bn) return;

    const float sx[4] = {0.0f, IMGF, IMGF, 0.0f};
    const float sy[4] = {0.0f, 0.0f, IMGF, IMGF};
    const int perm[8] = {0, 1, 2, 3, 6, 7, 4, 5};

    float off[8];
    for (int i = 0; i < 8; ++i) {
        float v = o1[b * 8 + perm[i]];
        if (o2) v += o2[b * 8 + perm[i]];
        off[i] = v * invScale;
    }

    float A[8][9];
    for (int i = 0; i < 4; ++i) {
        const float x = sx[i] * invScale;
        const float y = sy[i] * invScale;
        const float u = x + off[2 * i + 0];
        const float v = y + off[2 * i + 1];
        float* r0 = A[2 * i + 0];
        float* r1 = A[2 * i + 1];
        r0[0] = x; r0[1] = y; r0[2] = 1.0f; r0[3] = 0.0f; r0[4] = 0.0f; r0[5] = 0.0f;
        r0[6] = -u * x; r0[7] = -u * y; r0[8] = u;
        r1[0] = 0.0f; r1[1] = 0.0f; r1[2] = 0.0f; r1[3] = x; r1[4] = y; r1[5] = 1.0f;
        r1[6] = -v * x; r1[7] = -v * y; r1[8] = v;
    }

    for (int col = 0; col < 8; ++col) {
        int piv = col;
        float best = fabsf(A[col][col]);
        for (int r = col + 1; r < 8; ++r) {
            float a = fabsf(A[r][col]);
            if (a > best) { best = a; piv = r; }
        }
        if (piv != col)
            for (int c = 0; c < 9; ++c) {
                float tt = A[col][c]; A[col][c] = A[piv][c]; A[piv][c] = tt;
            }
        float dd = A[col][col];
        if (dd == 0.0f) dd = 1e-12f;
        const float inv = 1.0f / dd;
        for (int c = col; c < 9; ++c) A[col][c] *= inv;
        for (int r = 0; r < 8; ++r) {
            if (r == col) continue;
            const float f = A[r][col];
            if (f != 0.0f)
                for (int c = col; c < 9; ++c) A[r][c] -= f * A[col][c];
        }
    }

    const float a = A[0][8], b1 = A[1][8], c = A[2][8];
    const float d = A[3][8], e  = A[4][8], f = A[5][8];
    const float g = A[6][8], h  = A[7][8], i9 = 1.0f;

    float det = a * (e * i9 - f * h) - b1 * (d * i9 - f * g) + c * (d * h - e * g);
    if (det == 0.0f) det = 1e-12f;
    const float id = 1.0f / det;
    float Hi[3][3];
    Hi[0][0] = (e * i9 - f * h) * id;  Hi[0][1] = (c * h - b1 * i9) * id;  Hi[0][2] = (b1 * f - c * e) * id;
    Hi[1][0] = (f * g - d * i9) * id;  Hi[1][1] = (a * i9 - c * g) * id;   Hi[1][2] = (c * d - a * f) * id;
    Hi[2][0] = (d * h - e * g) * id;   Hi[2][1] = (b1 * g - a * h) * id;   Hi[2][2] = (a * e - b1 * d) * id;

    const float hs = size * 0.5f;
    float T1[3][3];
    for (int r = 0; r < 3; ++r) {
        T1[r][0] = Hi[r][0] * hs;
        T1[r][1] = Hi[r][1] * hs;
        T1[r][2] = Hi[r][0] * hs + Hi[r][1] * hs + Hi[r][2];
    }
    const float is = 2.0f / size;
    float* th = theta + b * 9;
    for (int cc = 0; cc < 3; ++cc) {
        th[0 * 3 + cc] = is * T1[0][cc] - T1[2][cc];
        th[1 * 3 + cc] = is * T1[1][cc] - T1[2][cc];
        th[2 * 3 + cc] = T1[2][cc];
    }
}

// ---------------------------------------------------------------------------
// Spatial transformer (bilinear, square same-size), NCHW in/out
// ---------------------------------------------------------------------------
__global__ void stn_k(const float* __restrict__ U, const float* __restrict__ theta,
                      float* __restrict__ out, int Bn, int C, int S)
{
    const int total = Bn * S * S;
    int t = blockIdx.x * blockDim.x + threadIdx.x;
    if (t >= total) return;
    const int x = t % S;
    const int y = (t / S) % S;
    const int b = t / (S * S);

    const float step = 2.0f / (float)(S - 1);
    const float xt = -1.0f + step * (float)x;
    const float yt = -1.0f + step * (float)y;

    const float* th = theta + b * 9;
    const float X = th[0] * xt + th[1] * yt + th[2];
    const float Y = th[3] * xt + th[4] * yt + th[5];
    float Z = th[6] * xt + th[7] * yt + th[8];
    Z = Z + 1e-6f * (1.0f - ((fabsf(Z) >= 1e-7f) ? 1.0f : 0.0f));

    const float xs = (X / Z + 1.0f) * (S * 0.5f);
    const float ys = (Y / Z + 1.0f) * (S * 0.5f);

    int x0 = (int)floorf(xs);
    int y0 = (int)floorf(ys);
    int x1 = x0 + 1, y1 = y0 + 1;
    x0 = min(max(x0, 0), S - 1); x1 = min(max(x1, 0), S - 1);
    y0 = min(max(y0, 0), S - 1); y1 = min(max(y1, 0), S - 1);

    const float x0f = (float)x0, x1f = (float)x1;
    const float y0f = (float)y0, y1f = (float)y1;
    const float wa = (x1f - xs) * (y1f - ys);
    const float wb = (x1f - xs) * (ys - y0f);
    const float wc = (xs - x0f) * (y1f - ys);
    const float wd = (xs - x0f) * (ys - y0f);

    const int SS = S * S;
    for (int c = 0; c < C; ++c) {
        const float* pc = U + ((long)(b * C + c)) * SS;
        const float Ia = pc[y0 * S + x0];
        const float Ib = pc[y1 * S + x0];
        const float Ic = pc[y0 * S + x1];
        const float Id = pc[y1 * S + x1];
        out[((long)(b * C + c)) * SS + y * S + x] = wa * Ia + wb * Ib + wc * Ic + wd * Id;
    }
}

// ---------------------------------------------------------------------------
// Final pack: out[b, s, j] = off_s[b, j]
// ---------------------------------------------------------------------------
__global__ void pack_out_k(const float* __restrict__ o1, const float* __restrict__ o2,
                           const float* __restrict__ o3, float* __restrict__ out, int Bn)
{
    int t = blockIdx.x * blockDim.x + threadIdx.x;
    if (t >= Bn * 24) return;
    const int b = t / 24;
    const int s = (t % 24) / 8;
    const int j = t % 8;
    const float* src = (s == 0) ? o1 : (s == 1) ? o2 : o3;
    out[t] = src[b * 8 + j];
}

// ---------------------------------------------------------------------------
// Host-side orchestration
// ---------------------------------------------------------------------------
extern "C" void kernel_launch(void* const* d_in, const int* in_sizes, int n_in,
                              void* d_out, int out_size, void* d_ws, size_t ws_size,
                              hipStream_t stream)
{
    (void)in_sizes; (void)n_in; (void)out_size; (void)ws_size;

    const int Bn = 4;
    const float* x1 = (const float*)d_in[0];
    const float* x2 = (const float*)d_in[1];

    // workspace carve (floats)
    float* ws = (float*)d_ws;
    size_t off = 0;
    auto alloc = [&](size_t n) { float* p = ws + off; off += n; return p; };
    float* t0    = alloc(4194304);  // ping   (max B*64*128*128)
    float* t1    = alloc(4194304);  // pong
    float* f1a   = alloc(1048576);  // B*64*64*64
    float* f2a   = alloc(1048576);
    float* f1b   = alloc(524288);   // B*128*32*32
    float* f2b   = alloc(524288);
    float* f1c   = alloc(131072);   // B*128*16*16
    float* f2c   = alloc(131072);
    float* n1    = alloc(1048576);  // normalized features (reused)
    float* n2    = alloc(1048576);
    float* corr  = alloc(1327104);  // max cost volume (B*81*64*64)
    float* w2b   = alloc(524288);
    float* w2a   = alloc(1048576);
    float* hid   = alloc(4096);     // B x 1024
    float* off1  = alloc(32);
    float* off2  = alloc(32);
    float* off3  = alloc(32);
    float* thet  = alloc(64);
    _Float16* Apack = (_Float16*)alloc(2520000); // max 32*307*512 = 5.03M f16
    _Float16* Bpack = (_Float16*)alloc(6040000); // max 1024*36*512 = 18.87M f16

    // Conv = pack weights once, then per-N-chunk: pack im2col + 2x2 GEMM.
    auto conv = [&](const float* in, int wi, float* out_, int Cin, int H, int W, int Cout) {
        const int K      = Cin * 9;
        const int KT     = (K + 31) / 32;
        const int Mtiles = (Cout + 15) / 16;     // always even here
        const int Npix   = Bn * H * W;
        const int HW     = H * W;
        const float* wp = (const float*)d_in[wi];
        const float* bp = (const float*)d_in[wi + 1];

        long wtot = (long)Mtiles * KT * 512;
        pack_w_k<<<dim3((unsigned)((wtot + 255) / 256)), 256, 0, stream>>>(
            wp, Apack, Mtiles, KT, Cout, K);

        const int totTiles   = Npix / 16;
        const int chunkTiles = 1024;             // 16384 pixels / chunk
        for (int tb = 0; tb < totTiles; tb += chunkTiles) {
            const int ct = (totTiles - tb < chunkTiles) ? (totTiles - tb) : chunkTiles;
            long btot = (long)ct * KT * 512;
            pack_b_k<<<dim3((unsigned)((btot + 255) / 256)), 256, 0, stream>>>(
                in, Bpack, ct, KT, tb * 16, Cin, H, W, K, Npix);
            dim3 g(Mtiles / 2, ct / 8);          // 2x2 macro-tiles, 4 waves/block
            conv_gemm_k<<<g, 128, 0, stream>>>(
                Apack, Bpack, bp, out_, KT, Cout, HW, Npix, tb * 16, 1);
        }
    };
    auto pool = [&](const float* in, float* out_, int C, int H, int W) {
        int total = Bn * C * (H / 2) * (W / 2);
        maxpool2_k<<<(total + 255) / 256, 256, 0, stream>>>(in, out_, Bn, C, H, W);
    };
    auto l2n = [&](const float* in, float* out_, int C, int HW) {
        int total = Bn * HW;
        l2norm_k<<<(total + 255) / 256, 256, 0, stream>>>(in, out_, Bn, C, HW);
    };
    auto fc = [&](const float* X, int wi, float* out_, long K, int M, int relu) {
        fc_wmma<<<(M + 63) / 64, 128, 0, stream>>>(
            X, (const float*)d_in[wi], (const float*)d_in[wi + 1], out_, Bn, K, M, relu);
    };

    // ---- feature pyramid for both images --------------------------------
    auto feat = [&](const float* x, float* fa, float* fb, float* fcv) {
        conv(x,  2, t0, 1, 128, 128, 64);
        conv(t0, 4, t1, 64, 128, 128, 64);
        pool(t1, t0, 64, 128, 128);
        conv(t0, 6, t1, 64, 64, 64, 64);
        conv(t1, 8, fa, 64, 64, 64, 64);          // f*a : B*64*64*64
        pool(fa, t0, 64, 64, 64);
        conv(t0, 10, t1, 64, 32, 32, 128);
        conv(t1, 12, fb, 128, 32, 32, 128);       // f*b : B*128*32*32
        pool(fb, t0, 128, 32, 32);
        conv(t0, 14, t1, 128, 16, 16, 128);
        conv(t1, 16, fcv, 128, 16, 16, 128);      // f*c : B*128*16*16
    };
    feat(x1, f1a, f1b, f1c);
    feat(x2, f2a, f2b, f2c);

    // ---- level 1: 16x16, sr=16 (1089 disp) ------------------------------
    l2n(f1c, n1, 128, 256);
    l2n(f2c, n2, 128, 256);
    {
        int total = Bn * 1089 * 256;
        corr_k<<<(total + 255) / 256, 256, 0, stream>>>(n1, n2, corr, Bn, 128, 16, 16, 16);
    }
    conv(corr, 18, t0, 1089, 16, 16, 512);
    conv(t0,   20, t1, 512, 16, 16, 512);
    conv(t1,   22, t0, 512, 16, 16, 512);
    fc(t0, 24, hid, 512L * 256, 1024, 1);
    fc(hid, 26, off1, 1024, 8, 0);

    // H2 = inv(dlt(src/4, off1/4)); theta2 = norm_H(H2, 32)
    dlt_norm_k<<<1, 32, 0, stream>>>(off1, nullptr, 0.25f, 32.0f, thet, Bn);
    {
        int total = Bn * 32 * 32;
        stn_k<<<(total + 255) / 256, 256, 0, stream>>>(f2b, thet, w2b, Bn, 128, 32);
    }

    // ---- level 2: 32x32, sr=8 (289 disp) --------------------------------
    l2n(f1b, n1, 128, 1024);
    l2n(w2b, n2, 128, 1024);
    {
        int total = Bn * 289 * 1024;
        corr_k<<<(total + 255) / 256, 256, 0, stream>>>(n1, n2, corr, Bn, 128, 32, 32, 8);
    }
    conv(corr, 28, t0, 289, 32, 32, 256);
    conv(t0,   30, t1, 256, 32, 32, 256);
    conv(t1,   32, t0, 256, 32, 32, 256);
    fc(t0, 34, hid, 256L * 1024, 1024, 1);
    fc(hid, 36, off2, 1024, 8, 0);

    // H3 = inv(dlt(src/2, (off1+off2)/2)); theta3 = norm_H(H3, 64)
    dlt_norm_k<<<1, 32, 0, stream>>>(off1, off2, 0.5f, 64.0f, thet, Bn);
    {
        int total = Bn * 64 * 64;
        stn_k<<<(total + 255) / 256, 256, 0, stream>>>(f2a, thet, w2a, Bn, 64, 64);
    }

    // ---- level 3: 64x64, sr=4 (81 disp) ---------------------------------
    l2n(f1a, n1, 64, 4096);
    l2n(w2a, n2, 64, 4096);
    {
        int total = Bn * 81 * 4096;
        corr_k<<<(total + 255) / 256, 256, 0, stream>>>(n1, n2, corr, Bn, 64, 64, 64, 4);
    }
    conv(corr, 38, t0, 81, 64, 64, 128);
    conv(t0,   40, t1, 128, 64, 64, 128);
    conv(t1,   42, t0, 128, 64, 64, 128);
    fc(t0, 44, hid, 128L * 4096, 1024, 1);
    fc(hid, 46, off3, 1024, 8, 0);

    // ---- pack (B, 3, 8) --------------------------------------------------
    pack_out_k<<<1, 128, 0, stream>>>(off1, off2, off3, (float*)d_out, Bn);
}
